// BiLSTM_CRF_37245956391034
// MI455X (gfx1250) — compile-verified, matrix operand order checked by reading, not run on hardware
//
#include <hip/hip_runtime.h>
#include <stdint.h>

typedef __attribute__((ext_vector_type(16))) __bf16 v16bf;
typedef __attribute__((ext_vector_type(8)))  float  v8f;

#define S_LEN   1024
#define E_DIM   512          // layer input width (E = 2H = 512 for both layers)
#define H_DIM   256
#define H4      1024         // 4*H
#define T_TAGS  12
#define START_TAG 10
#define STOP_TAG  11

__device__ __forceinline__ unsigned short f2bf(float f) {
  unsigned u = __float_as_uint(f);
  u += 0x7FFFu + ((u >> 16) & 1u);          // round-to-nearest-even
  return (unsigned short)(u >> 16);
}
__device__ __forceinline__ float bf2f(unsigned short s) {
  return __uint_as_float(((unsigned)s) << 16);
}

// ---------------- embedding gather: x[t][e] = emb[X[t]][e] ----------------
__global__ void k_gather(const int* __restrict__ X, const float* __restrict__ emb,
                         float* __restrict__ x) {
  int tid = blockIdx.x * blockDim.x + threadIdx.x;   // S*E threads
  int t = tid >> 9;
  int e = tid & 511;
  x[tid] = emb[(size_t)X[t] * E_DIM + e];
}

// ---------------- bias = b_ih + b_hh (both dirs of one layer) ----------------
__global__ void k_bias(const float* __restrict__ bi, const float* __restrict__ bh,
                       float* __restrict__ bias) {
  int tid = blockIdx.x * blockDim.x + threadIdx.x;   // 2*H4 threads
  bias[tid] = bi[tid] + bh[tid];
}

// ---------------- plain f32 -> bf16 convert (w_hh) ----------------
__global__ void k_cvt(const float* __restrict__ src, unsigned short* __restrict__ dst) {
  int tid = blockIdx.x * blockDim.x + threadIdx.x;
  dst[tid] = f2bf(src[tid]);
}

// -------- pack a row-major [R,K] f32 matrix into WMMA 16x32 bf16 tile layout --------
// tile (rt,kt): 512 bf16, index = lane*16 + e ; element maps to
//   row = rt*16 + lane%16 ;  K_local = e + 8*(lane>=16) + 8*(e>=8)
__global__ void k_pack(const float* __restrict__ src, unsigned short* __restrict__ dst,
                       int K) {
  int tid = blockIdx.x * blockDim.x + threadIdx.x;   // R*K threads
  int tile = tid >> 9;
  int rem  = tid & 511;
  int l = rem >> 4;
  int e = rem & 15;
  int ktiles = K >> 5;
  int rt = tile / ktiles;
  int kt = tile - rt * ktiles;
  int row = rt * 16 + (l & 15);
  int kl  = e + ((l >= 16) ? 8 : 0) + ((e >= 8) ? 8 : 0);
  dst[tid] = f2bf(src[(size_t)row * K + kt * 32 + kl]);
}

// ---------------- WMMA GEMM: G[d][t][n] = x @ w_ih[d]^T + bias[d] ----------------
// Block = 8 waves, computes a 32x256 output macro-tile (2 mt x 16 nt).
// Packed A slice for the block (32 KB, contiguous) is staged into LDS with
// gfx1250 global_load_async_to_lds_b128 (ASYNCcnt), then each wave runs
// 1x4 register blocking: one LDS A operand feeds 4 WMMAs on global B tiles.
__global__ void __launch_bounds__(256) k_gemm(
    const unsigned short* __restrict__ xb,
    const unsigned short* __restrict__ wb,
    const float* __restrict__ bias,
    float* __restrict__ G) {
  __shared__ unsigned short Ash[2 * 16 * 512];   // 32 KB: 2 mt x 16 kt x 512

  int d  = blockIdx.x >> 7;          // 128 blocks per dir
  int rm = blockIdx.x & 127;
  int mb = rm >> 2;                  // 0..31 -> mt_base = 2*mb
  int nb = rm & 3;                   // 0..3  -> nt_base = 16*nb
  int mt_base = mb * 2;
  int nt_base = nb * 16;

  // ---- async stage of packed A (contiguous 32 KB) into LDS ----
  {
    unsigned ldsBase = (unsigned)(uintptr_t)(&Ash[0]);
    const char* gsrc = (const char*)(xb + (size_t)mt_base * 16 * 512);
    int tid = threadIdx.x;
#pragma unroll
    for (int i = 0; i < 8; ++i) {
      int q = tid + i * 256;                 // uint4 index, 0..2047
      unsigned    laddr = ldsBase + (unsigned)(q * 16);
      const void* gaddr = gsrc + (size_t)q * 16;
      asm volatile("global_load_async_to_lds_b128 %0, %1, off"
                   :: "v"(laddr), "v"(gaddr) : "memory");
    }
    asm volatile("s_wait_asynccnt 0x0" ::: "memory");
  }
  __syncthreads();

  int w    = threadIdx.x >> 5;       // wave 0..7
  int lane = threadIdx.x & 31;
  int mt_local = w & 1;
  int nt0 = nt_base + (w >> 1) * 4;

  const unsigned short* B = wb + (size_t)d * (H4 * E_DIM);

  v8f acc0 = {}, acc1 = {}, acc2 = {}, acc3 = {};
#pragma unroll 2
  for (int kt = 0; kt < 16; ++kt) {
    v16bf a = *(const v16bf*)(&Ash[(size_t)(mt_local * 16 + kt) * 512 + lane * 16]);
    v16bf b0 = *(const v16bf*)(B + ((size_t)((nt0 + 0) * 16 + kt) * 512 + lane * 16));
    v16bf b1 = *(const v16bf*)(B + ((size_t)((nt0 + 1) * 16 + kt) * 512 + lane * 16));
    v16bf b2 = *(const v16bf*)(B + ((size_t)((nt0 + 2) * 16 + kt) * 512 + lane * 16));
    v16bf b3 = *(const v16bf*)(B + ((size_t)((nt0 + 3) * 16 + kt) * 512 + lane * 16));
    acc0 = __builtin_amdgcn_wmma_f32_16x16x32_bf16(false, a, false, b0, (short)0, acc0, false, false);
    acc1 = __builtin_amdgcn_wmma_f32_16x16x32_bf16(false, a, false, b1, (short)0, acc1, false, false);
    acc2 = __builtin_amdgcn_wmma_f32_16x16x32_bf16(false, a, false, b2, (short)0, acc2, false, false);
    acc3 = __builtin_amdgcn_wmma_f32_16x16x32_bf16(false, a, false, b3, (short)0, acc3, false, false);
  }

  int   mbase = (mt_base + mt_local) * 16 + ((lane >= 16) ? 8 : 0);
  float* Gd = G + (size_t)d * S_LEN * H4;
  v8f accs[4] = {acc0, acc1, acc2, acc3};
#pragma unroll
  for (int i = 0; i < 4; ++i) {
    int   n  = (nt0 + i) * 16 + (lane & 15);
    float bv = bias[d * H4 + n];
#pragma unroll
    for (int r = 0; r < 8; ++r)
      Gd[(size_t)(mbase + r) * H4 + n] = accs[i][r] + bv;
  }
}

// ---------------- sequential LSTM recurrence (one block per direction) ----------------
__device__ __forceinline__ float dot8(uint4 w, const float* h) {
  float s;
  s  = bf2f((unsigned short)(w.x & 0xFFFF)) * h[0];
  s += bf2f((unsigned short)(w.x >> 16))    * h[1];
  s += bf2f((unsigned short)(w.y & 0xFFFF)) * h[2];
  s += bf2f((unsigned short)(w.y >> 16))    * h[3];
  s += bf2f((unsigned short)(w.z & 0xFFFF)) * h[4];
  s += bf2f((unsigned short)(w.z >> 16))    * h[5];
  s += bf2f((unsigned short)(w.w & 0xFFFF)) * h[6];
  s += bf2f((unsigned short)(w.w >> 16))    * h[7];
  return s;
}

__global__ void k_lstm(const float* __restrict__ G, const unsigned short* __restrict__ whb,
                       const float* __restrict__ h0, const float* __restrict__ c0,
                       float* __restrict__ xout) {
  int d = blockIdx.x;
  int j = threadIdx.x;                 // 256 threads, one per hidden unit
  __shared__ float hsh[H_DIM];

  const float*          Gd = G   + (size_t)d * S_LEN * H4;
  const unsigned short* W  = whb + (size_t)d * H4 * H_DIM;
  const unsigned short* Wi = W + (size_t)(j)       * H_DIM;
  const unsigned short* Wf = W + (size_t)(j + 256) * H_DIM;
  const unsigned short* Wg = W + (size_t)(j + 512) * H_DIM;
  const unsigned short* Wo = W + (size_t)(j + 768) * H_DIM;

  float c = c0[d * H_DIM + j];
  hsh[j]  = h0[d * H_DIM + j];
  __syncthreads();

  for (int s = 0; s < S_LEN; ++s) {
    int t = d ? (S_LEN - 1 - s) : s;
    const float* g0 = Gd + (size_t)t * H4;
    float ai = g0[j], af = g0[256 + j], ag = g0[512 + j], ao = g0[768 + j];
    if (s + 1 < S_LEN) {               // gfx1250 global_prefetch_b8 of next step's gates
      int tn = d ? (t - 1) : (t + 1);
      __builtin_prefetch(Gd + (size_t)tn * H4 + j, 0, 1);
    }
#pragma unroll 4
    for (int k = 0; k < H_DIM; k += 8) {
      const float* hp = &hsh[k];
      ai += dot8(*(const uint4*)(Wi + k), hp);
      af += dot8(*(const uint4*)(Wf + k), hp);
      ag += dot8(*(const uint4*)(Wg + k), hp);
      ao += dot8(*(const uint4*)(Wo + k), hp);
    }
    float si = 1.f / (1.f + __expf(-ai));
    float sf = 1.f / (1.f + __expf(-af));
    float so = 1.f / (1.f + __expf(-ao));
    c = sf * c + si * tanhf(ag);
    float h = so * tanhf(c);
    xout[(size_t)t * (2 * H_DIM) + d * H_DIM + j] = h;
    __syncthreads();
    hsh[j] = h;
    __syncthreads();
  }
}

// ---------------- output projection: feats[t][tag] ----------------
__global__ void k_feats(const float* __restrict__ x, const float* __restrict__ w_out,
                        const float* __restrict__ b_out, float* __restrict__ feats) {
  int tid = blockIdx.x * blockDim.x + threadIdx.x;    // S*T threads
  if (tid >= S_LEN * T_TAGS) return;
  int t = tid / T_TAGS;
  int tag = tid - t * T_TAGS;
  const float* xr = x + (size_t)t * (2 * H_DIM);
  const float* wr = w_out + (size_t)tag * (2 * H_DIM);
  float s = b_out[tag];
#pragma unroll 4
  for (int k = 0; k < 2 * H_DIM; ++k) s += xr[k] * wr[k];
  feats[tid] = s;
}

// ---------------- Viterbi decode (single wave; backpointers in LDS) ----------------
__global__ void k_viterbi(const float* __restrict__ feats, const float* __restrict__ trans,
                          float* __restrict__ score_out, float* __restrict__ path_out) {
  __shared__ float trs[T_TAGS * T_TAGS];
  __shared__ float fv[T_TAGS];
  __shared__ float term[T_TAGS];
  __shared__ int   bps[S_LEN * T_TAGS];     // 48 KB of the 320 KB WGP LDS
  int j = threadIdx.x;
  for (int i = j; i < T_TAGS * T_TAGS; i += blockDim.x) trs[i] = trans[i];
  if (j < T_TAGS) fv[j] = (j == START_TAG) ? 0.f : -10000.f;
  __syncthreads();

  for (int t = 0; t < S_LEN; ++t) {
    float best = -3.0e38f; int bp = 0;
    if (j < T_TAGS) {
      for (int p = 0; p < T_TAGS; ++p) {
        float sc = fv[p] + trs[j * T_TAGS + p];
        if (sc > best) { best = sc; bp = p; }
      }
    }
    __syncthreads();
    if (j < T_TAGS) {
      fv[j] = best + feats[t * T_TAGS + j];
      bps[t * T_TAGS + j] = bp;
    }
    __syncthreads();
  }
  if (j < T_TAGS) term[j] = fv[j] + trs[STOP_TAG * T_TAGS + j];
  __syncthreads();
  if (j == 0) {
    int bl = 0; float bs = term[0];
    for (int p = 1; p < T_TAGS; ++p) if (term[p] > bs) { bs = term[p]; bl = p; }
    score_out[0] = bs;
    int tag = bl;
    for (int t = S_LEN - 1; t >= 0; --t) {
      path_out[t] = (float)tag;
      tag = bps[t * T_TAGS + tag];
    }
  }
}

extern "C" void kernel_launch(void* const* d_in, const int* in_sizes, int n_in,
                              void* d_out, int out_size, void* d_ws, size_t ws_size,
                              hipStream_t stream) {
  (void)in_sizes; (void)n_in; (void)out_size; (void)ws_size;
  const int*   X     = (const int*)d_in[0];
  const float* emb   = (const float*)d_in[1];
  const float* w_ih  = (const float*)d_in[2];   // [L,2,4H,2H]
  const float* w_hh  = (const float*)d_in[3];   // [L,2,4H,H]
  const float* b_ih  = (const float*)d_in[4];   // [L,2,4H]
  const float* b_hh  = (const float*)d_in[5];
  const float* h0    = (const float*)d_in[6];   // [L,2,H]
  const float* c0    = (const float*)d_in[7];
  const float* w_out = (const float*)d_in[8];   // [T,2H]
  const float* b_out = (const float*)d_in[9];
  const float* trans = (const float*)d_in[10];  // [T,T]
  float* out = (float*)d_out;

  char* ws = (char*)d_ws;
  float*          x    = (float*)(ws + 0);                       // 2 MB  [S,512]
  unsigned short* xb   = (unsigned short*)(ws + (2u  << 20));    // 1 MB  packed activations
  unsigned short* wb   = (unsigned short*)(ws + (3u  << 20));    // 2 MB  packed w_ih (2 dirs)
  float*          bias = (float*)(ws + (5u  << 20));             // 8 KB  [2,4H]
  float*          G    = (float*)(ws + (6u  << 20));             // 8 MB  [2,S,4H]
  unsigned short* whb  = (unsigned short*)(ws + (14u << 20));    // 1 MB  bf16 w_hh (2 dirs)

  k_gather<<<(S_LEN * E_DIM) / 256, 256, 0, stream>>>(X, emb, x);

  for (int l = 0; l < 2; ++l) {
    k_bias<<<(2 * H4) / 256, 256, 0, stream>>>(b_ih + (size_t)l * 2 * H4,
                                               b_hh + (size_t)l * 2 * H4, bias);
    for (int d = 0; d < 2; ++d)
      k_pack<<<(H4 * E_DIM) / 256, 256, 0, stream>>>(
          w_ih + (size_t)(l * 2 + d) * H4 * E_DIM,
          wb + (size_t)d * H4 * E_DIM, E_DIM);
    k_cvt<<<(2 * H4 * H_DIM) / 256, 256, 0, stream>>>(
        w_hh + (size_t)l * 2 * H4 * H_DIM, whb);
    k_pack<<<(S_LEN * E_DIM) / 256, 256, 0, stream>>>(x, xb, E_DIM);
    k_gemm<<<256, 256, 0, stream>>>(xb, wb, bias, G);
    k_lstm<<<2, 256, 0, stream>>>(G, whb, h0 + (size_t)l * 2 * H_DIM,
                                  c0 + (size_t)l * 2 * H_DIM, x);
  }

  k_feats<<<48, 256, 0, stream>>>(x, w_out, b_out, out);
  k_viterbi<<<1, 32, 0, stream>>>(out, trans,
                                  out + S_LEN * T_TAGS,
                                  out + S_LEN * T_TAGS + 1);
}
